// DETM_53085795778858
// MI455X (gfx1250) — matrix operand order repaired and checked by reading, not compiled, not run on hardware
//
#include <hip/hip_runtime.h>
#include <math.h>

// ---------------- problem constants ----------------
constexpr int NK   = 50;     // topics
constexpr int NT   = 100;    // timesteps
constexpr int NV   = 10000;  // vocab
constexpr int NTH  = 800;    // theta hidden
constexpr int NEH  = 256;    // eta hidden
constexpr int NRHO = 300;    // rho dim
constexpr int NB   = 256;    // batch
#define LOGD (-5.298317366548036f)   /* log(0.005) */

typedef __attribute__((ext_vector_type(16))) __bf16 v16bf;
typedef __attribute__((ext_vector_type(8)))  float  v8f;
typedef __attribute__((ext_vector_type(2)))  int    v2i_t;
typedef __bf16 bf16_t;

// -------- CDNA5 async global->LDS staging (ASYNCcnt), feature-guarded --------
#if defined(__has_builtin)
#if __has_builtin(__builtin_amdgcn_global_load_async_to_lds_b64) && \
    __has_builtin(__builtin_amdgcn_s_wait_asynccnt)
#define HAVE_ASYNC_LDS 1
#endif
#endif
#ifndef HAVE_ASYNC_LDS
#define HAVE_ASYNC_LDS 0
#endif

__device__ __forceinline__ void copy8_g2l(const float* g, float* l) {
#if HAVE_ASYNC_LDS
  __builtin_amdgcn_global_load_async_to_lds_b64(
      (__attribute__((address_space(1))) v2i_t*)g,
      (__attribute__((address_space(3))) v2i_t*)l, 0, 0);
#else
  *(float2*)l = *(const float2*)g;
#endif
}
__device__ __forceinline__ void wait_async_then_barrier() {
#if HAVE_ASYNC_LDS
  __builtin_amdgcn_s_wait_asynccnt(0);
#endif
  __syncthreads();
}

// ---------------- workspace layout (floats) ----------------
// acc[0]=kl_alpha acc[1]=kl_eta acc[2]=kl_theta_raw acc[3]=nll_raw
constexpr size_t OFF_ACC   = 0;
constexpr size_t OFF_X0    = 16;                      // [100,256]
constexpr size_t OFF_X1    = OFF_X0 + 100 * 256;      // [100,256]
constexpr size_t OFF_ETA   = OFF_X1 + 100 * 256;      // [100,50]
constexpr size_t OFF_H1    = OFF_ETA + 100 * 50;      // [256,800]
constexpr size_t OFF_H2    = OFF_H1 + 256 * 800;      // [256,800]
constexpr size_t OFF_MUT   = OFF_H2 + 256 * 800;      // [256,50]
constexpr size_t OFF_LST   = OFF_MUT + 256 * 50;      // [256,50]
constexpr size_t OFF_THETA = OFF_LST + 256 * 50;      // [256,50]
constexpr size_t OFF_LT    = OFF_THETA + 256 * 50;    // [50,10000] logits tile
constexpr size_t OFF_RMAX  = OFF_LT + 50 * 10000;     // [64]
constexpr size_t OFF_RSUM  = OFF_RMAX + 64;           // [64]  (stores 1/sum)

__device__ __forceinline__ float sigmoidf_(float x) { return 1.0f / (1.0f + expf(-x)); }

__device__ float block_reduce_sum(float v, float* sbuf) {
  int tid = threadIdx.x;
  sbuf[tid] = v;
  __syncthreads();
  for (int s = blockDim.x >> 1; s > 0; s >>= 1) {
    if (tid < s) sbuf[tid] += sbuf[tid + s];
    __syncthreads();
  }
  return sbuf[0];
}

// ---------------- fragment helpers (ISA 7.12.2 layouts, f32 LDS -> bf16) ------
__device__ __forceinline__ v16bf load_afrag(const float (*sA)[32], int row, int half) {
  v16bf a;
  for (int j = 0; j < 8; j++) {
    a[j]     = (bf16_t)sA[row][half * 8 + j];
    a[8 + j] = (bf16_t)sA[row][16 + half * 8 + j];
  }
  return a;
}
__device__ __forceinline__ v16bf load_bfrag(const float (*sB)[32], int row, int half) {
  v16bf b;
  for (int j = 0; j < 8; j++) {
    b[j]     = (bf16_t)sB[row][half * 8 + j];
    b[8 + j] = (bf16_t)sB[row][16 + half * 8 + j];
  }
  return b;
}

// ---------------- dense WMMA GEMM: C[M,N] = A[M,K] @ Bw[N,K]^T ----------------
// mode: bit0 = relu, bit1 = accumulate into C
// block tile 64x64, 8 wave32 waves: wave (wr,wc) -> rows [wr*32,+32) x cols [wc*16,+16)
__global__ void __launch_bounds__(256) wmma_gemm_dense(
    const float* __restrict__ A, long lda,
    const float* __restrict__ Bw, long ldb,
    const float* __restrict__ bias, float* __restrict__ C,
    int M, int N, int Kd, int ldc, int mode) {
  __shared__ float sA[64][32];
  __shared__ float sB[64][32];
  const int tid  = threadIdx.x;
  const int lane = tid & 31;
  const int wave = tid >> 5;
  const int wr   = wave >> 2;   // 0..1
  const int wc   = wave & 3;    // 0..3
  const int half = lane >> 4;   // 0..1
  const int l15  = lane & 15;
  const int blockM = blockIdx.y * 64;
  const int blockN = blockIdx.x * 64;
  const bool fullM = (blockM + 64) <= M;
  const bool fullN = (blockN + 64) <= N;

  v8f acc0, acc1;
  for (int i = 0; i < 8; i++) { acc0[i] = 0.0f; acc1[i] = 0.0f; }

  for (int k0 = 0; k0 < Kd; k0 += 32) {
    const bool fullK = (k0 + 32) <= Kd;
    // stage 64x32 f32 tiles (1024 float2 slots each), async when in-bounds
    for (int e = tid; e < 1024; e += 256) {
      int r = e >> 4, c2 = (e & 15) << 1;
      int gk = k0 + c2;
      {
        int gm = blockM + r;
        if (fullK && fullM) {
          copy8_g2l(&A[(long)gm * lda + gk], &sA[r][c2]);
        } else {
          sA[r][c2]     = (gm < M && gk < Kd)     ? A[(long)gm * lda + gk]     : 0.0f;
          sA[r][c2 + 1] = (gm < M && gk + 1 < Kd) ? A[(long)gm * lda + gk + 1] : 0.0f;
        }
      }
      {
        int gn = blockN + r;
        if (fullK && fullN) {
          copy8_g2l(&Bw[(long)gn * ldb + gk], &sB[r][c2]);
        } else {
          sB[r][c2]     = (gn < N && gk < Kd)     ? Bw[(long)gn * ldb + gk]     : 0.0f;
          sB[r][c2 + 1] = (gn < N && gk + 1 < Kd) ? Bw[(long)gn * ldb + gk + 1] : 0.0f;
        }
      }
    }
    wait_async_then_barrier();

    v16bf bfrag = load_bfrag(sB, wc * 16 + l15, half);
    v16bf afrag = load_afrag(sA, wr * 32 + l15, half);
    acc0 = __builtin_amdgcn_wmma_f32_16x16x32_bf16(false, afrag, false, bfrag,
                                                   (short)0, acc0, false, false);
    afrag = load_afrag(sA, wr * 32 + 16 + l15, half);
    acc1 = __builtin_amdgcn_wmma_f32_16x16x32_bf16(false, afrag, false, bfrag,
                                                   (short)0, acc1, false, false);
    __syncthreads();
  }

  int n = blockN + wc * 16 + l15;
  if (n < N) {
    float bv = bias ? bias[n] : 0.0f;
    for (int i = 0; i < 8; i++) {
      int m0 = blockM + wr * 32 + i + 8 * half;
      int m1 = m0 + 16;
      if (m0 < M) {
        float v = acc0[i] + bv;
        if (mode & 2) v += C[(long)m0 * ldc + n];
        if (mode & 1) v = v > 0.0f ? v : 0.0f;
        C[(long)m0 * ldc + n] = v;
      }
      if (m1 < M) {
        float v = acc1[i] + bv;
        if (mode & 2) v += C[(long)m1 * ldc + n];
        if (mode & 1) v = v > 0.0f ? v : 0.0f;
        C[(long)m1 * ldc + n] = v;
      }
    }
  }
}

// ---------------- generic guarded WMMA GEMM (small / gathered A) --------------
struct DenseA {
  const float* p; int ld;
  __device__ float get(int m, int k) const { return p[(long)m * ld + k]; }
};
struct EtaGatherA {  // A[m,k] = eta[times[m], k], K = NK
  const float* eta; const int* times;
  __device__ float get(int m, int k) const { return eta[times[m] * NK + k]; }
};

template <typename AL>
__global__ void __launch_bounds__(256) wmma_gemm_gen(AL A, const float* __restrict__ Bw,
                                                     const float* __restrict__ bias,
                                                     float* __restrict__ C,
                                                     int M, int N, int Kd, int ldb, int ldc,
                                                     int mode) {
  __shared__ float sA[64][32];
  __shared__ float sB[64][32];
  const int tid  = threadIdx.x;
  const int lane = tid & 31;
  const int wave = tid >> 5;
  const int wr   = wave >> 2;
  const int wc   = wave & 3;
  const int half = lane >> 4;
  const int l15  = lane & 15;
  const int blockM = blockIdx.y * 64;
  const int blockN = blockIdx.x * 64;

  v8f acc0, acc1;
  for (int i = 0; i < 8; i++) { acc0[i] = 0.0f; acc1[i] = 0.0f; }

  for (int k0 = 0; k0 < Kd; k0 += 32) {
    for (int e = tid; e < 64 * 32; e += 256) {
      int r = e >> 5, c = e & 31;
      int gm = blockM + r, gk = k0 + c;
      sA[r][c] = (gm < M && gk < Kd) ? A.get(gm, gk) : 0.0f;
      int gn = blockN + r;
      sB[r][c] = (gn < N && gk < Kd) ? Bw[(long)gn * ldb + gk] : 0.0f;
    }
    __syncthreads();
    v16bf bfrag = load_bfrag(sB, wc * 16 + l15, half);
    v16bf afrag = load_afrag(sA, wr * 32 + l15, half);
    acc0 = __builtin_amdgcn_wmma_f32_16x16x32_bf16(false, afrag, false, bfrag,
                                                   (short)0, acc0, false, false);
    afrag = load_afrag(sA, wr * 32 + 16 + l15, half);
    acc1 = __builtin_amdgcn_wmma_f32_16x16x32_bf16(false, afrag, false, bfrag,
                                                   (short)0, acc1, false, false);
    __syncthreads();
  }

  int n = blockN + wc * 16 + l15;
  if (n < N) {
    float bv = bias ? bias[n] : 0.0f;
    for (int i = 0; i < 8; i++) {
      int m0 = blockM + wr * 32 + i + 8 * half;
      int m1 = m0 + 16;
      if (m0 < M) {
        float v = acc0[i] + bv;
        if (mode & 2) v += C[(long)m0 * ldc + n];
        if (mode & 1) v = v > 0.0f ? v : 0.0f;
        C[(long)m0 * ldc + n] = v;
      }
      if (m1 < M) {
        float v = acc1[i] + bv;
        if (mode & 2) v += C[(long)m1 * ldc + n];
        if (mode & 1) v = v > 0.0f ? v : 0.0f;
        C[(long)m1 * ldc + n] = v;
      }
    }
  }
}

// ---------------- small kernels ----------------
__global__ void zero_acc_kernel(float* ws) {
  if (threadIdx.x < 16) ws[threadIdx.x] = 0.0f;
}

__global__ void kl_alpha_kernel(const float* __restrict__ mqa,
                                const float* __restrict__ lsa, float* acc) {
  __shared__ float sred[256];
  const long n = (long)NK * NT * NRHO;
  float local = 0.0f;
  for (long e = blockIdx.x * (long)blockDim.x + threadIdx.x; e < n;
       e += (long)gridDim.x * blockDim.x) {
    long kt = e / NRHO;
    int  t  = (int)(kt % NT);
    float qmu = mqa[e], qls = lsa[e];
    float pmu = (t == 0) ? 0.0f : mqa[e - NRHO];
    float pls = (t == 0) ? 0.0f : LOGD;
    float d = qmu - pmu;
    local += 0.5f * ((expf(qls) + d * d) / (expf(pls) + 1e-6f) - 1.0f + pls - qls);
  }
  float tot = block_reduce_sum(local, sred);
  if (threadIdx.x == 0) atomicAdd(acc + 0, tot);
}

__global__ void __launch_bounds__(1024) lstm_kernel(const float* __restrict__ Wih,
                                                    const float* __restrict__ Whh,
                                                    const float* __restrict__ bih,
                                                    const float* __restrict__ bhh,
                                                    const float* __restrict__ xin,
                                                    float* __restrict__ xout) {
  __shared__ float sx[NEH], sh[NEH], sc[NEH], sg[4 * NEH];
  int tid = threadIdx.x;
  if (tid < NEH) { sh[tid] = 0.0f; sc[tid] = 0.0f; }
  __syncthreads();
  for (int t = 0; t < NT; t++) {
    if (tid < NEH) sx[tid] = xin[t * NEH + tid];
    __syncthreads();
    float g = bih[tid] + bhh[tid];
    const float* wi = Wih + (long)tid * NEH;
    const float* wh = Whh + (long)tid * NEH;
    for (int kk = 0; kk < NEH; kk++) g += wi[kk] * sx[kk] + wh[kk] * sh[kk];
    sg[tid] = g;
    __syncthreads();
    if (tid < NEH) {
      float gi = sg[tid], gf = sg[NEH + tid], gg = sg[2 * NEH + tid], go = sg[3 * NEH + tid];
      float c = sigmoidf_(gf) * sc[tid] + sigmoidf_(gi) * tanhf(gg);
      float h = sigmoidf_(go) * tanhf(c);
      sc[tid] = c; sh[tid] = h;
      xout[t * NEH + tid] = h;
    }
    __syncthreads();
  }
}

__global__ void __launch_bounds__(128) eta_kernel(const float* __restrict__ x3,
                                                  const float* __restrict__ Wm,
                                                  const float* __restrict__ bm,
                                                  const float* __restrict__ Wl,
                                                  const float* __restrict__ bl,
                                                  float* __restrict__ eta_out,
                                                  float* acc) {
  __shared__ float sprev[NK], smu[NK], sls[NK], sred[128];
  int tid = threadIdx.x;
  if (tid < NK) sprev[tid] = 0.0f;
  float klacc = 0.0f;
  __syncthreads();
  for (int t = 0; t < NT; t++) {
    if (tid < NK) {
      const float* w = Wm + (long)tid * (NEH + NK);
      float d = bm[tid];
      for (int j = 0; j < NEH; j++) d += w[j] * x3[t * NEH + j];
      for (int j = 0; j < NK; j++) d += w[NEH + j] * sprev[j];
      smu[tid] = d;
    } else if (tid >= 64 && tid < 64 + NK) {
      int k = tid - 64;
      const float* w = Wl + (long)k * (NEH + NK);
      float d = bl[k];
      for (int j = 0; j < NEH; j++) d += w[j] * x3[t * NEH + j];
      for (int j = 0; j < NK; j++) d += w[NEH + j] * sprev[j];
      sls[k] = d;
    }
    __syncthreads();
    if (tid < NK) {
      float qmu = smu[tid], qls = sls[tid];
      float pmu = (t == 0) ? 0.0f : sprev[tid];
      float pls = (t == 0) ? 0.0f : LOGD;
      float d = qmu - pmu;
      klacc += 0.5f * ((expf(qls) + d * d) / (expf(pls) + 1e-6f) - 1.0f + pls - qls);
    }
    __syncthreads();
    if (tid < NK) {
      sprev[tid] = smu[tid];
      eta_out[t * NK + tid] = smu[tid];
    }
    __syncthreads();
  }
  float tot = block_reduce_sum(klacc, sred);
  if (tid == 0) atomicAdd(acc + 1, tot);
}

__global__ void __launch_bounds__(64) theta_kernel(const float* __restrict__ mu_t,
                                                   const float* __restrict__ ls_t,
                                                   const float* __restrict__ eta,
                                                   const int* __restrict__ times,
                                                   float* __restrict__ theta_out,
                                                   float* acc) {
  __shared__ float smu[NK], sls2[NK], se[NK], sred[64];
  __shared__ float smx, ssum;
  int b = blockIdx.x, tid = threadIdx.x;
  if (tid < NK) {
    smu[tid]  = mu_t[b * NK + tid];
    sls2[tid] = ls_t[b * NK + tid];
    se[tid]   = eta[times[b] * NK + tid];
  }
  __syncthreads();
  if (tid == 0) {
    float mx = -3.4e38f;
    for (int k = 0; k < NK; k++) mx = fmaxf(mx, smu[k]);
    float s = 0.0f;
    for (int k = 0; k < NK; k++) s += expf(smu[k] - mx);
    smx = mx; ssum = s;
  }
  __syncthreads();
  float local = 0.0f;
  if (tid < NK) {
    theta_out[b * NK + tid] = expf(smu[tid] - smx) / ssum;
    float d = smu[tid] - se[tid];
    local = 0.5f * ((expf(sls2[tid]) + d * d) / (1.0f + 1e-6f) - 1.0f - sls2[tid]);
  }
  float tot = block_reduce_sum(local, sred);
  if (tid == 0) atomicAdd(acc + 2, tot);
}

__global__ void __launch_bounds__(256) stats_kernel(const float* __restrict__ Lt,
                                                    float* __restrict__ rmax,
                                                    float* __restrict__ rsum) {
  __shared__ float sred[256];
  __shared__ float smx;
  int k = blockIdx.x, tid = threadIdx.x;
  const float* row = Lt + (long)k * NV;
  float mx = -3.4e38f;
  for (int v = tid; v < NV; v += 256) mx = fmaxf(mx, row[v]);
  sred[tid] = mx;
  __syncthreads();
  for (int s = 128; s > 0; s >>= 1) {
    if (tid < s) sred[tid] = fmaxf(sred[tid], sred[tid + s]);
    __syncthreads();
  }
  if (tid == 0) smx = sred[0];
  __syncthreads();
  float sum = 0.0f;
  for (int v = tid; v < NV; v += 256) sum += expf(row[v] - smx);
  float tot = block_reduce_sum(sum, sred);
  if (tid == 0) { rmax[k] = smx; rsum[k] = 1.0f / tot; }
}

__global__ void __launch_bounds__(256) nll_kernel(const float* __restrict__ Lt,
                                                  const float* __restrict__ rmax,
                                                  const float* __restrict__ rsuminv,
                                                  const float* __restrict__ theta,
                                                  const float* __restrict__ bows,
                                                  const int* __restrict__ times,
                                                  int t, float* acc) {
  __shared__ float sred[256];
  __shared__ float smx[NK], sinv[NK];
  __shared__ int   slist[NB];
  __shared__ int   scnt;
  int tid = threadIdx.x;
  if (tid == 0) scnt = 0;
  if (tid < NK) { smx[tid] = rmax[tid]; sinv[tid] = rsuminv[tid]; }
  __syncthreads();
  if (tid < NB && times[tid] == t) {
    int pos = atomicAdd(&scnt, 1);
    slist[pos] = tid;
  }
  __syncthreads();
  int v = blockIdx.x * 256 + tid;
  float local = 0.0f;
  if (v < NV) {
    for (int ii = 0; ii < scnt; ii++) {
      int b = slist[ii];
      float s = 0.0f;
      for (int k = 0; k < NK; k++)
        s += theta[b * NK + k] * expf(Lt[(long)k * NV + v] - smx[k]) * sinv[k];
      local += bows[(long)b * NV + v] * (-logf(s + 1e-6f));
    }
  }
  float tot = block_reduce_sum(local, sred);
  if (tid == 0) atomicAdd(acc + 3, tot);
}

__global__ void finalize_kernel(const float* __restrict__ acc,
                                const int* __restrict__ num_docs,
                                float* __restrict__ out) {
  if (threadIdx.x == 0 && blockIdx.x == 0) {
    float coeff = (float)(*num_docs) / (float)NB;
    float kla = acc[0], kle = acc[1];
    float klt = acc[2] * coeff;
    float nll = acc[3] * coeff;
    out[0] = nll + kla + kle + klt;
    out[1] = nll;
    out[2] = kla;
    out[3] = kle;
    out[4] = klt;
  }
}

// ---------------- launcher ----------------
extern "C" void kernel_launch(void* const* d_in, const int* in_sizes, int n_in,
                              void* d_out, int out_size, void* d_ws, size_t ws_size,
                              hipStream_t stream) {
  (void)in_sizes; (void)n_in; (void)out_size; (void)ws_size;

  const float* bows      = (const float*)d_in[0];
  const float* nbows     = (const float*)d_in[1];
  const int*   times     = (const int*)d_in[2];
  const float* rnn_inp   = (const float*)d_in[3];
  const int*   num_docs  = (const int*)d_in[4];
  const float* mu_q_a    = (const float*)d_in[5];
  const float* ls_q_a    = (const float*)d_in[6];
  const float* rho_W     = (const float*)d_in[7];
  const float* qt_W1     = (const float*)d_in[8];
  const float* qt_b1     = (const float*)d_in[9];
  const float* qt_W2     = (const float*)d_in[10];
  const float* qt_b2     = (const float*)d_in[11];
  const float* mu_qt_W   = (const float*)d_in[12];
  const float* mu_qt_b   = (const float*)d_in[13];
  const float* ls_qt_W   = (const float*)d_in[14];
  const float* ls_qt_b   = (const float*)d_in[15];
  const float* qe_map_W  = (const float*)d_in[16];
  const float* qe_map_b  = (const float*)d_in[17];
  const float* lstm_w[3][4];
  for (int l = 0; l < 3; l++)
    for (int j = 0; j < 4; j++) lstm_w[l][j] = (const float*)d_in[18 + 4 * l + j];
  const float* mu_qe_W = (const float*)d_in[30];
  const float* mu_qe_b = (const float*)d_in[31];
  const float* ls_qe_W = (const float*)d_in[32];
  const float* ls_qe_b = (const float*)d_in[33];

  float* ws    = (float*)d_ws;
  float* acc   = ws + OFF_ACC;
  float* x0    = ws + OFF_X0;
  float* x1    = ws + OFF_X1;
  float* eta   = ws + OFF_ETA;
  float* h1    = ws + OFF_H1;
  float* h2    = ws + OFF_H2;
  float* mu_t  = ws + OFF_MUT;
  float* ls_t  = ws + OFF_LST;
  float* theta = ws + OFF_THETA;
  float* Lt    = ws + OFF_LT;
  float* rmax  = ws + OFF_RMAX;
  float* rsum  = ws + OFF_RSUM;
  float* out   = (float*)d_out;

  zero_acc_kernel<<<1, 32, 0, stream>>>(ws);

  // kl_alpha
  kl_alpha_kernel<<<1024, 256, 0, stream>>>(mu_q_a, ls_q_a, acc);

  // eta path: x0 = rnn_inp @ qe_map_W^T + b   (M=100, N=256, K=10000)
  wmma_gemm_dense<<<dim3(4, 2), 256, 0, stream>>>(
      rnn_inp, (long)NV, qe_map_W, (long)NV, qe_map_b, x0, NT, NEH, NV, NEH, 0);
  // LSTM: x0 -> x1 -> x0 -> x1
  lstm_kernel<<<1, 1024, 0, stream>>>(lstm_w[0][0], lstm_w[0][1], lstm_w[0][2], lstm_w[0][3], x0, x1);
  lstm_kernel<<<1, 1024, 0, stream>>>(lstm_w[1][0], lstm_w[1][1], lstm_w[1][2], lstm_w[1][3], x1, x0);
  lstm_kernel<<<1, 1024, 0, stream>>>(lstm_w[2][0], lstm_w[2][1], lstm_w[2][2], lstm_w[2][3], x0, x1);
  eta_kernel<<<1, 128, 0, stream>>>(x1, mu_qe_W, mu_qe_b, ls_qe_W, ls_qe_b, eta, acc);

  // theta path: h1 = relu(nbows @ W1a^T + eta_td @ W1b^T + b1), split dense+gather
  wmma_gemm_dense<<<dim3(13, 4), 256, 0, stream>>>(
      nbows, (long)NV, qt_W1, (long)(NV + NK), qt_b1, h1, NB, NTH, NV, NTH, /*store+bias*/0);
  wmma_gemm_gen<EtaGatherA><<<dim3(13, 4), 256, 0, stream>>>(
      EtaGatherA{eta, times}, qt_W1 + NV, nullptr, h1, NB, NTH, NK, NV + NK, NTH, /*accum+relu*/3);
  wmma_gemm_dense<<<dim3(13, 4), 256, 0, stream>>>(
      h1, (long)NTH, qt_W2, (long)NTH, qt_b2, h2, NB, NTH, NTH, NTH, /*relu*/1);
  wmma_gemm_gen<DenseA><<<dim3(1, 4), 256, 0, stream>>>(
      DenseA{h2, NTH}, mu_qt_W, mu_qt_b, mu_t, NB, NK, NTH, NTH, NK, 0);
  wmma_gemm_gen<DenseA><<<dim3(1, 4), 256, 0, stream>>>(
      DenseA{h2, NTH}, ls_qt_W, ls_qt_b, ls_t, NB, NK, NTH, NTH, NK, 0);
  theta_kernel<<<NB, 64, 0, stream>>>(mu_t, ls_t, eta, times, theta, acc);

  // beta + nll, looped over t with a reused 50x10000 logits tile (L2-hot)
  // alpha[t] is dense: A = mu_q_alpha + t*RHO with lda = T*RHO
  for (int t = 0; t < NT; t++) {
    wmma_gemm_dense<<<dim3(157, 1), 256, 0, stream>>>(
        mu_q_a + (long)t * NRHO, (long)NT * NRHO, rho_W, (long)NRHO, nullptr, Lt,
        NK, NV, NRHO, NV, 0);
    stats_kernel<<<NK, 256, 0, stream>>>(Lt, rmax, rsum);
    nll_kernel<<<40, 256, 0, stream>>>(Lt, rmax, rsum, theta, bows, times, t, acc);
  }

  finalize_kernel<<<1, 32, 0, stream>>>(acc, num_docs, out);
}
